// BertSelfAttention_40235253629188
// MI455X (gfx1250) — matrix operand order repaired and checked by reading, not compiled, not run on hardware
//
#include <hip/hip_runtime.h>
#include <hip/hip_bf16.h>

#define HDIM 1024
#define NH   16
#define HD   64
#define BB   2
#define LL   2048

typedef __attribute__((ext_vector_type(16))) _Float16 v16h;
typedef __attribute__((ext_vector_type(8)))  _Float16 v8h;
typedef __attribute__((ext_vector_type(8)))  float    v8f;
typedef __attribute__((ext_vector_type(4)))  float    v4f;

__device__ __forceinline__ v8f wmma16(v16h a, v16h b, v8f c) {
    return __builtin_amdgcn_wmma_f32_16x16x32_f16(false, a, false, b, (short)0, c, false, false);
}

// 8 halves from p0 -> elements 0..7, 8 halves from p1 -> elements 8..15
__device__ __forceinline__ v16h load16h(const _Float16* __restrict__ p0,
                                        const _Float16* __restrict__ p1) {
    v16h r;
#pragma unroll
    for (int i = 0; i < 8; ++i) { r[i] = p0[i]; r[8 + i] = p1[i]; }
    return r;
}

// ---------------------------------------------------------------------------
// Kernel 0: one-shot fp32 -> f16 conversion (X and the three weight matrices
// get converted once instead of on every GEMM k-step).
// ---------------------------------------------------------------------------
__global__ __launch_bounds__(256) void cvt_f16_kernel(const float* __restrict__ src,
                                                      _Float16* __restrict__ dst,
                                                      int n8) {
    const int i = blockIdx.x * blockDim.x + threadIdx.x;
    if (i >= n8) return;
    const v4f a = *(const v4f*)(src + (size_t)i * 8);
    const v4f b = *(const v4f*)(src + (size_t)i * 8 + 4);
    v8h r;
#pragma unroll
    for (int j = 0; j < 4; ++j) { r[j] = (_Float16)a[j]; r[4 + j] = (_Float16)b[j]; }
    *(v8h*)(dst + (size_t)i * 8) = r;
}

// ---------------------------------------------------------------------------
// Kernel 1: QKV projection on pre-converted f16 operands.
// out = X @ W.T + b for Wq/Wk/Wv.  One wave computes a 32(M) x 64(N) tile
// (8 independent accumulator chains -> hides WMMA RAW hazard, 2x B reuse).
// Q,K stored f16 [B, NH, L, HD]; V stored f16 transposed [B, NH, HD, L].
// ---------------------------------------------------------------------------
__global__ __launch_bounds__(256) void qkv_proj_kernel(
        const _Float16* __restrict__ Xh,
        const _Float16* __restrict__ Wqh, const float* __restrict__ bq,
        const _Float16* __restrict__ Wkh, const float* __restrict__ bk,
        const _Float16* __restrict__ Wvh, const float* __restrict__ bv,
        _Float16* __restrict__ Qh, _Float16* __restrict__ Kh,
        _Float16* __restrict__ VTh)
{
    const int wave = (blockIdx.x * blockDim.x + threadIdx.x) >> 5;   // 0..6143
    const int lane = threadIdx.x & 31;
    const int li = lane & 15;
    const int hi = lane >> 4;

    const int mat = wave / (128 * 16);          // 0=Q,1=K,2=V
    const int rem = wave % (128 * 16);
    const int m0  = (rem / 16) * 32;            // token tile base
    const int n0  = (rem % 16) * 64;            // output-feature tile base

    const _Float16* __restrict__ Wm = (mat == 0) ? Wqh : ((mat == 1) ? Wkh : Wvh);
    const float*    __restrict__ bm = (mat == 0) ? bq  : ((mat == 1) ? bk  : bv);

    v8f acc[2][4];
#pragma unroll
    for (int s = 0; s < 2; ++s)
#pragma unroll
        for (int t = 0; t < 4; ++t) acc[s][t] = v8f{};

    const _Float16* __restrict__ xr0 = Xh + (size_t)(m0 + li) * HDIM;
    const _Float16* __restrict__ xr1 = Xh + (size_t)(m0 + 16 + li) * HDIM;
    const int offA = hi * 16;   // A-layout k sub-offset
    const int offB = hi * 16;   // B-layout k sub-offset

#pragma unroll 2
    for (int ko = 0; ko < HDIM; ko += 32) {
        const v16h a0 = load16h(xr0 + ko + offA, xr0 + ko + 8 + offA);
        const v16h a1 = load16h(xr1 + ko + offA, xr1 + ko + 8 + offA);
#pragma unroll
        for (int t = 0; t < 4; ++t) {
            // B operand column n = row (n0+16t+li) of W, contiguous along k.
            const v16h bop = *(const v16h*)(Wm + (size_t)(n0 + 16 * t + li) * HDIM + ko + offB);
            acc[0][t] = wmma16(a0, bop, acc[0][t]);
            acc[1][t] = wmma16(a1, bop, acc[1][t]);
        }
    }

    // Store with bias. C/D layout: lane holds col n=li, vgpr r -> row r + 8*hi.
#pragma unroll
    for (int t = 0; t < 4; ++t) {
        const int n = n0 + 16 * t + li;
        const float bias = bm[n];
        const int h = n >> 6;
        const int d = n & 63;
#pragma unroll
        for (int s = 0; s < 2; ++s) {
#pragma unroll
            for (int r = 0; r < 8; ++r) {
                const int tok  = m0 + 16 * s + r + hi * 8;
                const int bidx = tok >> 11;
                const int lpos = tok & (LL - 1);
                const _Float16 hv = (_Float16)(acc[s][t][r] + bias);
                const size_t head = (size_t)(bidx * NH + h);
                if (mat == 0)      Qh [(head * LL + lpos) * HD + d] = hv;
                else if (mat == 1) Kh [(head * LL + lpos) * HD + d] = hv;
                else               VTh[(head * HD + d) * LL + lpos] = hv;
            }
        }
    }
}

// ---------------------------------------------------------------------------
// Kernel 2: flash attention. One wave per (b, h, 16-query tile).
// Streams keys in 32-key steps with online softmax; f32 accumulation.
// Row-sums of P are computed by the matrix pipe (P x ones) instead of
// cross-lane shuffle reductions; only the max reduction stays on VALU.
// ---------------------------------------------------------------------------
__global__ __launch_bounds__(128) void attn_kernel(
        const _Float16* __restrict__ Qh, const _Float16* __restrict__ Kh,
        const _Float16* __restrict__ VTh, const float* __restrict__ mask,
        float* __restrict__ out)
{
    __shared__ _Float16 lds[4][16 * 32];      // per-wave P staging (1 KB each)

    const int wslot = threadIdx.x >> 5;
    const int lane  = threadIdx.x & 31;
    const int li = lane & 15;
    const int hi = lane >> 4;

    const int wave = blockIdx.x * 4 + wslot;  // 0..4095
    const int q0 = (wave & 127) * 16;
    const int bh = wave >> 7;                 // 0..31
    const int b  = bh >> 4;
    const int h  = bh & 15;

    const _Float16* __restrict__ Qb = Qh  + (size_t)bh * LL * HD;
    const _Float16* __restrict__ Kb = Kh  + (size_t)bh * LL * HD;
    const _Float16* __restrict__ Vb = VTh + (size_t)bh * HD * LL;
    const float*    __restrict__ mr = mask + (size_t)b * LL;

    // Cache Q A-operand (16 queries x 64 dims -> two 16x32 A tiles)
    const _Float16* qrow = Qb + (size_t)(q0 + li) * HD;
    const v16h qa0 = load16h(qrow +  0 + hi * 16, qrow +  8 + hi * 16);
    const v16h qa1 = load16h(qrow + 32 + hi * 16, qrow + 40 + hi * 16);

    // all-ones B operand: acc_s = P x ones gives per-row sums of P in every
    // column -> same C/D layout as the other accumulators, no redistribution.
    v16h ones;
#pragma unroll
    for (int i = 0; i < 16; ++i) ones[i] = (_Float16)1.0f;

    v8f acc[4] = {v8f{}, v8f{}, v8f{}, v8f{}};
    v8f acc_s = {};                            // running softmax denominators
    float mrun[8];
#pragma unroll
    for (int r = 0; r < 8; ++r) mrun[r] = -1e30f;

    _Float16* __restrict__ P = &lds[wslot][0];
    const _Float16* __restrict__ prow = P + li * 32;   // A-layout read base
    const float sc = 0.125f;                           // 1/sqrt(64)

    for (int kt = 0; kt < LL; kt += 32) {
        // --- scores: two 16x16 tiles over this 32-key strip -------------
        const _Float16* kr0 = Kb + (size_t)(kt + li) * HD + hi * 16;
        const _Float16* kr1 = Kb + (size_t)(kt + 16 + li) * HD + hi * 16;
        v8f s0 = {}, s1 = {};
        // interleaved so dependent WMMAs are separated by an independent one
        s0 = wmma16(qa0, *(const v16h*)(kr0),      s0);
        s1 = wmma16(qa0, *(const v16h*)(kr1),      s1);
        s0 = wmma16(qa1, *(const v16h*)(kr0 + 32), s0);
        s1 = wmma16(qa1, *(const v16h*)(kr1 + 32), s1);

        // prefetch next K strip (global_prefetch_b8)
        if (kt + 32 < LL) {
            __builtin_prefetch(Kb + (size_t)(kt + 32 + li) * HD + hi * 16, 0, 3);
            __builtin_prefetch(Kb + (size_t)(kt + 48 + li) * HD + hi * 16, 0, 3);
        }

        const float mk0 = mr[kt + li];
        const float mk1 = mr[kt + 16 + li];

        // --- online softmax: rows live per-vgpr, cols across 16-lane half
#pragma unroll
        for (int r = 0; r < 8; ++r) {
            const float a = s0[r] * sc + mk0;
            const float c = s1[r] * sc + mk1;
            float mx = fmaxf(a, c);
            mx = fmaxf(mx, __shfl_xor(mx, 1, 32));
            mx = fmaxf(mx, __shfl_xor(mx, 2, 32));
            mx = fmaxf(mx, __shfl_xor(mx, 4, 32));
            mx = fmaxf(mx, __shfl_xor(mx, 8, 32));
            const float mnew = fmaxf(mrun[r], mx);
            const float alpha = __expf(mrun[r] - mnew);
            mrun[r] = mnew;
            const float e0 = __expf(a - mnew);
            const float e1 = __expf(c - mnew);
            acc[0][r] *= alpha; acc[1][r] *= alpha;
            acc[2][r] *= alpha; acc[3][r] *= alpha;
            acc_s[r]  *= alpha;
            // stage P (C/D layout -> row-major 16x32 in LDS)
            const int row = r + hi * 8;
            P[row * 32 + li]      = (_Float16)e0;
            P[row * 32 + 16 + li] = (_Float16)e1;
        }

        // Same-wave LDS ops are in-order; compiler inserts s_wait_dscnt for RAW.
        const v16h pa = load16h(prow + hi * 16, prow + 8 + hi * 16);

        // --- context: acc += P (16x32) x V (32keys x 64dims) ------------
        acc_s = wmma16(pa, ones, acc_s);       // row-sums via the matrix pipe
#pragma unroll
        for (int t = 0; t < 4; ++t) {
            const _Float16* vr = Vb + (size_t)(16 * t + li) * LL + kt + hi * 16;
            acc[t] = wmma16(pa, *(const v16h*)vr, acc[t]);
        }
    }

    // --- normalize and store fp32 to out[b][q][h*64 + d] ----------------
    float rdenom[8];
#pragma unroll
    for (int r = 0; r < 8; ++r) rdenom[r] = 1.0f / acc_s[r];
#pragma unroll
    for (int t = 0; t < 4; ++t) {
#pragma unroll
        for (int r = 0; r < 8; ++r) {
            const int qidx = q0 + r + hi * 8;
            const int d    = 16 * t + li;
            out[((size_t)b * LL + qidx) * HDIM + h * HD + d] = acc[t][r] * rdenom[r];
        }
    }
}

// ---------------------------------------------------------------------------
extern "C" void kernel_launch(void* const* d_in, const int* in_sizes, int n_in,
                              void* d_out, int out_size, void* d_ws, size_t ws_size,
                              hipStream_t stream) {
    const float* X    = (const float*)d_in[0];
    const float* mask = (const float*)d_in[1];
    const float* Wq   = (const float*)d_in[2];
    const float* bq   = (const float*)d_in[3];
    const float* Wk   = (const float*)d_in[4];
    const float* bk   = (const float*)d_in[5];
    const float* Wv   = (const float*)d_in[6];
    const float* bv   = (const float*)d_in[7];
    float* out = (float*)d_out;

    const size_t perQKV = (size_t)BB * NH * LL * HD;  // 4,194,304 halves
    const size_t nX = (size_t)BB * LL * HDIM;         // 4,194,304
    const size_t nW = (size_t)HDIM * HDIM;            // 1,048,576

    _Float16* Qh  = (_Float16*)d_ws;
    _Float16* Kh  = Qh  + perQKV;
    _Float16* VTh = Kh  + perQKV;
    _Float16* Xh  = VTh + perQKV;
    _Float16* Wqh = Xh  + nX;
    _Float16* Wkh = Wqh + nW;
    _Float16* Wvh = Wkh + nW;

    // --- pre-pass: fp32 -> f16 once --------------------------------------
    cvt_f16_kernel<<<(int)(nX / 8 / 256), 256, 0, stream>>>(X,  Xh,  (int)(nX / 8));
    cvt_f16_kernel<<<(int)(nW / 8 / 256), 256, 0, stream>>>(Wq, Wqh, (int)(nW / 8));
    cvt_f16_kernel<<<(int)(nW / 8 / 256), 256, 0, stream>>>(Wk, Wkh, (int)(nW / 8));
    cvt_f16_kernel<<<(int)(nW / 8 / 256), 256, 0, stream>>>(Wv, Wvh, (int)(nW / 8));

    // --- projection: 3 mats * 128 m-tiles * 16 n-groups = 6144 waves ------
    qkv_proj_kernel<<<768, 256, 0, stream>>>(Xh, Wqh, bq, Wkh, bk, Wvh, bv, Qh, Kh, VTh);

    // --- attention: 2*16*128 = 4096 waves, 4 waves/block ------------------
    attn_kernel<<<1024, 128, 0, stream>>>(Qh, Kh, VTh, mask, out);
}
